// AffinityNet_25623774888269
// MI455X (gfx1250) — matrix operand order repaired and checked by reading, not compiled, not run on hardware
//
#include <hip/hip_runtime.h>

// ---------------------------------------------------------------------------
// Types for CDNA5 WMMA
// ---------------------------------------------------------------------------
typedef __attribute__((ext_vector_type(16))) __bf16 v16bf;
typedef __attribute__((ext_vector_type(4)))  __bf16 v4bf;
typedef __attribute__((ext_vector_type(8)))  float  v8f;
typedef __attribute__((ext_vector_type(4)))  unsigned int v4u;

union AFragU { v16bf v; v4u q[2]; };

// ---------------------------------------------------------------------------
// GEMM:  OUT[b, chan_off+m, n] = elu( sum_k W[m,k] * X[b,k,n] )
//   W: [M,K] fp32 row-major, X: [B,K,N] fp32, OUT: [B,out_ch,N] fp32.
// Block tile 64(M) x 128(N) x 32(K), 256 threads = 8 wave32s in a 2x4 grid;
// each wave computes a 32x32 patch = 4 x v_wmma_f32_16x16x32_bf16 per K-step
// from 2 A-fragments and 2 B-fragments.
// Double-buffered LDS: one barrier per K-step; global loads for tile i+1
// overlap WMMA on tile i, and its LDS stores land in the idle buffer.
// K,N are template params so tile loads lower to immediate-offset vmem ops.
// ---------------------------------------------------------------------------
#define TMB 64
#define TNB 128
#define TKB 32
#define LDP 40   // LDS row pitch in bf16 (multiple of 8 -> 16B-aligned chunks)

template<int K, int N>
__global__ __launch_bounds__(256)
void gemm_elu(const float* __restrict__ W, const float* __restrict__ X,
              float* __restrict__ OUT, int out_ch, int chan_off)
{
    __shared__ __bf16 lA[2][TMB][LDP];   // A tile [m][k], double-buffered
    __shared__ __bf16 lB[2][TNB][LDP];   // B tile transposed [n][k]

    const int tid  = threadIdx.x;
    const int wave = tid >> 5;              // wave32
    const int lane = tid & 31;
    const int half = lane >> 4;
    const int lr   = lane & 15;

    const int b  = blockIdx.z;
    const int m0 = blockIdx.y * TMB;
    const int n0 = blockIdx.x * TNB;
    const int wm = wave >> 2;               // 0..1
    const int wn = wave & 3;                // 0..3

    const float* Xb = X + (size_t)b * K * N;

    // ---- per-thread staging assignments (fixed) ----
    // A: 64x32 = 2048 fp32, 2 float4 per thread
    const int ar0 = tid >> 3;               // rows 0..31
    const int ak0 = (tid & 7) * 4;          // k chunk
    const int ar1 = ar0 + 32;               // rows 32..63
    // B: 32(k) x 128(n), thread owns column nj, 16 consecutive k
    const int nj = tid & 127;
    const int kg = (tid >> 7) * 16;         // 0 or 16
    const float* aptr0 = W + (size_t)(m0 + ar0) * K + ak0;
    const float* aptr1 = W + (size_t)(m0 + ar1) * K + ak0;
    const float* bptr  = Xb + (size_t)kg * N + n0 + nj;

    float4 areg[2];
    float  breg[16];

    auto load_tiles = [&](int kk) {
        areg[0] = *(const float4*)(aptr0 + kk);
        areg[1] = *(const float4*)(aptr1 + kk);
        const float* bc = bptr + (size_t)kk * N;
        #pragma unroll
        for (int i = 0; i < 16; ++i) breg[i] = bc[i * N];   // immediate offsets
        if (kk + 2 * TKB < K)
            __builtin_prefetch(bc + 2 * (size_t)TKB * N, 0, 1);
    };
    auto store_tiles = [&](int buf) {
        v4bf a0 = {(__bf16)areg[0].x, (__bf16)areg[0].y, (__bf16)areg[0].z, (__bf16)areg[0].w};
        v4bf a1 = {(__bf16)areg[1].x, (__bf16)areg[1].y, (__bf16)areg[1].z, (__bf16)areg[1].w};
        *(v4bf*)&lA[buf][ar0][ak0] = a0;
        *(v4bf*)&lA[buf][ar1][ak0] = a1;
        #pragma unroll
        for (int c = 0; c < 4; ++c) {
            v4bf t = {(__bf16)breg[4*c], (__bf16)breg[4*c+1],
                      (__bf16)breg[4*c+2], (__bf16)breg[4*c+3]};
            *(v4bf*)&lB[buf][nj][kg + 4*c] = t;
        }
    };

    v8f acc[2][2] = {};

    // Prologue: stage tile 0 into buffer 0.
    load_tiles(0);
    store_tiles(0);
    __syncthreads();

    int cur = 0;
    for (int kk = 0; kk < K; kk += TKB) {
        const bool more = (kk + TKB) < K;
        if (more) load_tiles(kk + TKB);     // global loads overlap WMMA below

        // Fragments (layouts per CDNA5 ISA 7.12.2):
        // A 16x32: lanes 0-15 -> K {0..7,16..23}; lanes 16-31 -> K {8..15,24..31}
        AFragU af[2], bfr[2];
        #pragma unroll
        for (int mi = 0; mi < 2; ++mi) {
            const int am = wm * 32 + mi * 16 + lr;
            af[mi].q[0] = *(const v4u*)&lA[cur][am][half * 8];
            af[mi].q[1] = *(const v4u*)&lA[cur][am][16 + half * 8];
        }
        // B 32x16 (stored [n][k]): lanes 0-15 -> K {0..7,16..23}; lanes 16-31 -> K {8..15,24..31}
        #pragma unroll
        for (int ni = 0; ni < 2; ++ni) {
            const int bn = wn * 32 + ni * 16 + lr;
            bfr[ni].q[0] = *(const v4u*)&lB[cur][bn][half * 8];
            bfr[ni].q[1] = *(const v4u*)&lB[cur][bn][16 + half * 8];
        }
        #pragma unroll
        for (int mi = 0; mi < 2; ++mi)
            #pragma unroll
            for (int ni = 0; ni < 2; ++ni)
                acc[mi][ni] = __builtin_amdgcn_wmma_f32_16x16x32_bf16(
                    false, af[mi].v, false, bfr[ni].v, (short)0, acc[mi][ni],
                    false, false);

        if (more) {
            store_tiles(cur ^ 1);           // fill the idle buffer
            __syncthreads();                // single barrier per K-step
        }
        cur ^= 1;
    }

    // ---- epilogue: ELU + store. C/D layout: VGPR r -> M = 8*half + r, N = lane%16
    float* outb = OUT + ((size_t)b * out_ch + chan_off) * N;
    #pragma unroll
    for (int mi = 0; mi < 2; ++mi) {
        const int mb = m0 + wm * 32 + mi * 16 + half * 8;
        #pragma unroll
        for (int ni = 0; ni < 2; ++ni) {
            const int n = n0 + wn * 32 + ni * 16 + lr;
            #pragma unroll
            for (int r = 0; r < 8; ++r) {
                float v = acc[mi][ni][r];
                v = (v > 0.0f) ? v : (__expf(v) - 1.0f);
                outb[(size_t)(mb + r) * N + n] = v;
            }
        }
    }
}

// ---------------------------------------------------------------------------
// Bilinear 64->32 (align_corners=False, scale 2) == 2x2 average pool.
// F2: [B,64,64,64] -> XCAT channels [0,64) of [B,512,32,32]
// ---------------------------------------------------------------------------
__global__ __launch_bounds__(256)
void pool_f2_kernel(const float* __restrict__ F2, float* __restrict__ XCAT)
{
    int i = blockIdx.x * 256 + threadIdx.x;      // over B*64*1024
    if (i >= 8 * 64 * 1024) return;
    int n = i & 1023;
    int c = (i >> 10) & 63;
    int b = i >> 16;
    int y = n >> 5, x = n & 31;
    const float* src = F2 + (size_t)(b * 64 + c) * 4096;
    int base = (2 * y) * 64 + 2 * x;
    float v = 0.25f * (src[base] + src[base + 1] + src[base + 64] + src[base + 65]);
    XCAT[(size_t)(b * 512 + c) * 1024 + n] = v;
}

// ---------------------------------------------------------------------------
// Affinity: aff[b,p,j] = exp(-mean_c |x[c,to(p,j)] - x[c,from(j)]|)
// from = r*32 + 4 + col ; to = (r+dy[p])*32 + 4 + dx[p] + col ; j = r*24+col
// ---------------------------------------------------------------------------
__constant__ int AFF_DY[34] = {0,0,0,0,
                               1,1,1,1,1,1,1,1,1,
                               2,2,2,2,2,2,2,2,2,
                               3,3,3,3,3,3,3,
                               4,4,4,4,4};
__constant__ int AFF_DX[34] = {1,2,3,4,
                               -4,-3,-2,-1,0,1,2,3,4,
                               -4,-3,-2,-1,0,1,2,3,4,
                               -3,-2,-1,0,1,2,3,
                               -2,-1,0,1,2};

__global__ __launch_bounds__(256)
void affinity_kernel(const float* __restrict__ X, float* __restrict__ out)
{
    int j = blockIdx.x * blockDim.x + threadIdx.x;   // 0..671
    int p = blockIdx.y;                              // 0..33
    int b = blockIdx.z;                              // 0..7
    if (j >= 672) return;

    int r   = j / 24;
    int col = j - r * 24;
    int from = r * 32 + 4 + col;
    int to   = (r + AFF_DY[p]) * 32 + 4 + AFF_DX[p] + col;

    const float* xf = X + (size_t)b * 512 * 1024 + from;
    const float* xt = X + (size_t)b * 512 * 1024 + to;
    float s = 0.0f;
    #pragma unroll 8
    for (int c = 0; c < 512; ++c) {
        s += fabsf(xt[c * 1024] - xf[c * 1024]);
    }
    out[((size_t)b * 34 + p) * 672 + j] = __expf(-s * (1.0f / 512.0f));
}

// ---------------------------------------------------------------------------
// Host launcher
// d_in: 0=f2_in [8,512,64,64], 1=f3_in [8,1024,32,32], 2=f4_in [8,2048,32,32],
//       3=w2 [64,512], 4=w3 [128,1024], 5=w4 [320,2048], 6=w9 [512,512]
// d_out: [8,34,672] fp32
// ---------------------------------------------------------------------------
extern "C" void kernel_launch(void* const* d_in, const int* in_sizes, int n_in,
                              void* d_out, int out_size, void* d_ws, size_t ws_size,
                              hipStream_t stream)
{
    const float* f2_in = (const float*)d_in[0];
    const float* f3_in = (const float*)d_in[1];
    const float* f4_in = (const float*)d_in[2];
    const float* w2    = (const float*)d_in[3];
    const float* w3    = (const float*)d_in[4];
    const float* w4    = (const float*)d_in[5];
    const float* w9    = (const float*)d_in[6];

    // Workspace layout (floats): xcat | xbuf | f2full  (~42 MB total)
    float* xcat   = (float*)d_ws;                          // [8,512,1024]
    float* xbuf   = xcat + (size_t)8 * 512 * 1024;         // [8,512,1024]
    float* f2full = xbuf + (size_t)8 * 512 * 1024;         // [8,64,4096]

    dim3 blk(256);

    // f2: [64x512] x [512x4096] per batch -> elu -> f2full
    gemm_elu<512, 4096><<<dim3(4096 / TNB, 64 / TMB, 8), blk, 0, stream>>>(
        w2, f2_in, f2full, 64, 0);

    // bilinear(64->32) == avg-pool 2x2 -> xcat channels [0,64)
    pool_f2_kernel<<<(8 * 64 * 1024 + 255) / 256, blk, 0, stream>>>(f2full, xcat);

    // f3: [128x1024] x [1024x1024] -> xcat channels [64,192)
    gemm_elu<1024, 1024><<<dim3(1024 / TNB, 128 / TMB, 8), blk, 0, stream>>>(
        w3, f3_in, xcat, 512, 64);

    // f4: [320x2048] x [2048x1024] -> xcat channels [192,512)
    gemm_elu<2048, 1024><<<dim3(1024 / TNB, 320 / TMB, 8), blk, 0, stream>>>(
        w4, f4_in, xcat, 512, 192);

    // w9: [512x512] x [512x1024] -> xbuf
    gemm_elu<512, 1024><<<dim3(1024 / TNB, 512 / TMB, 8), blk, 0, stream>>>(
        w9, xcat, xbuf, 512, 0);

    // affinity -> d_out [8,34,672]
    affinity_kernel<<<dim3(3, 34, 8), blk, 0, stream>>>(xbuf, (float*)d_out);
}
